// SimpleGRU_45810121179425
// MI455X (gfx1250) — matrix operand order repaired and checked by reading, not compiled, not run on hardware
//
#include <hip/hip_runtime.h>
#include <hip/hip_bf16.h>

typedef __attribute__((ext_vector_type(16))) _Float16 v16h;
typedef __attribute__((ext_vector_type(8)))  float    v8f;

#define B_  32
#define T_  2048
#define H_  256
#define G_  768   // 2H (r,z) + H (n)

// ---------------- ws layout (bytes) ----------------
// [0)        Wx_f16 : 768*256 f16  rows 0..511 = W_irz, 512..767 = W_in
// [393216)   Wh_f16 : 768*256 f16  rows 0..511 = W_hrz, 512..767 = W_hn
// [786432)   bias_x : 768 f32  (b_irz || b_in)
// [789504)   bias_h : 768 f32  (b_hrz || b_hn)
// [792576)   xproj  : (B*T) x 768 f32   (x-side projections + bias_x)

__device__ __forceinline__ v8f wmma_f16(v16h a, v16h b, v8f c) {
  // D = A(16x32,f16) * B(32x16,f16) + C(16x16,f32)
  return __builtin_amdgcn_wmma_f32_16x16x32_f16(
      /*neg_a=*/false, a, /*neg_b=*/false, b,
      /*c_mod=*/(short)0, c, /*reuse_a=*/false, /*reuse_b=*/false);
}

// Build A-fragment for one lane: 16 fp32 at base[0..15] -> elems 0..15
// (contiguous k order, matching B).
__device__ __forceinline__ v16h make_a_frag(const float* base) {
  float4 p0 = ((const float4*)base)[0];
  float4 p1 = ((const float4*)base)[1];
  float4 p2 = ((const float4*)(base + 8))[0];
  float4 p3 = ((const float4*)(base + 8))[1];
  v16h a;
  a[0]=(_Float16)p0.x;  a[1]=(_Float16)p0.y;  a[2]=(_Float16)p0.z;  a[3]=(_Float16)p0.w;
  a[4]=(_Float16)p1.x;  a[5]=(_Float16)p1.y;  a[6]=(_Float16)p1.z;  a[7]=(_Float16)p1.w;
  a[8]=(_Float16)p2.x;  a[9]=(_Float16)p2.y;  a[10]=(_Float16)p2.z; a[11]=(_Float16)p2.w;
  a[12]=(_Float16)p3.x; a[13]=(_Float16)p3.y; a[14]=(_Float16)p3.z; a[15]=(_Float16)p3.w;
  return a;
}

__device__ __forceinline__ float fast_sigmoid(float x) {
  return __builtin_amdgcn_rcpf(1.0f + __expf(-x));      // v_rcp_f32, ~1 ulp
}
__device__ __forceinline__ float fast_tanh(float x) {
  // tanh(x) = 1 - 2/(exp(2x)+1); saturates correctly at +/-inf.
  return 1.0f - 2.0f * __builtin_amdgcn_rcpf(__expf(2.0f * x) + 1.0f);
}

// -------- kernel 1: pack weights to f16, fuse biases --------
__global__ __launch_bounds__(256)
void gru_pack(const float* __restrict__ Wirz, const float* __restrict__ Win,
              const float* __restrict__ Whrz, const float* __restrict__ Whn,
              const float* __restrict__ birz, const float* __restrict__ bin,
              const float* __restrict__ bhrz, const float* __restrict__ bhn,
              _Float16* __restrict__ Wx, _Float16* __restrict__ Wh,
              float* __restrict__ bx, float* __restrict__ bh) {
  const int g = blockIdx.x;          // 0..767 output-gate row
  const int k = threadIdx.x;         // 0..255 input column
  const float wx = (g < 512) ? Wirz[g * 256 + k] : Win[(g - 512) * 256 + k];
  const float wh = (g < 512) ? Whrz[g * 256 + k] : Whn[(g - 512) * 256 + k];
  Wx[g * 256 + k] = (_Float16)wx;
  Wh[g * 256 + k] = (_Float16)wh;
  if (k == 0) {
    bx[g] = (g < 512) ? birz[g] : bin[g - 512];
    bh[g] = (g < 512) ? bhrz[g] : bhn[g - 512];
  }
}

// -------- kernel 2: xproj = x @ [W_irz;W_in]^T + [b_irz;b_in] --------
// M = B*T = 65536, N = 768, K = 256. One block = one 16-row M-tile,
// 8 waves, each wave does six 16-col N-tiles. A-fragments for the whole
// K dim live in registers (64 VGPRs); compiler rotates to k-outer and
// overlaps B-load clauses with the WMMA chains.
__global__ __launch_bounds__(256)
void gru_xproj(const float* __restrict__ x, const _Float16* __restrict__ Wx,
               const float* __restrict__ bx, float* __restrict__ xproj) {
  const int lane = threadIdx.x & 31;
  const int wave = threadIdx.x >> 5;        // 0..7
  const int lc   = lane & 15;
  const bool hi  = lane >= 16;
  const int m0   = blockIdx.x * 16;         // M-tile base
  const int n0   = wave * 96;               // 6 N-tiles
  const int koff = hi ? 16 : 0;

  const float* xrow = x + (size_t)(m0 + lc) * 256;
  v16h a[8];
#pragma unroll
  for (int kb = 0; kb < 8; ++kb)
    a[kb] = make_a_frag(xrow + kb * 32 + (hi ? 16 : 0));

  v8f acc[6] = {};
#pragma unroll
  for (int j = 0; j < 6; ++j) {
    const _Float16* wrow = Wx + (size_t)(n0 + j * 16 + lc) * 256 + koff;
#pragma unroll
    for (int kb = 0; kb < 8; ++kb) {
      v16h b = *(const v16h*)(wrow + kb * 32);
      acc[j] = wmma_f16(a[kb], b, acc[j]);
    }
  }
#pragma unroll
  for (int j = 0; j < 6; ++j) {
    const int nb = n0 + j * 16;
    const float bias = bx[nb + lc];
#pragma unroll
    for (int vr = 0; vr < 8; ++vr) {
      const int row = m0 + vr + (hi ? 8 : 0);
      xproj[(size_t)row * G_ + nb + lc] = acc[j][vr] + bias;
    }
  }
}

// -------- kernel 3: sequential GRU scan --------
// 2 blocks (16 batch rows each) x 16 waves. Wave w owns gate N-tiles
// {w, w+16, w+32} = r/z/n for hidden columns [16w, 16w+16): all three
// gates for a column live in one wave -> no cross-wave gate exchange.
// All 24 recurrent B-fragments (192 VGPRs) are register-resident for the
// whole 2048-step loop; h is double-buffered in LDS (1 barrier/step);
// xproj for step t+1 is prefetched (global_prefetch_b8: no VGPR cost)
// while step t's WMMAs run, so demand loads hit near caches.
__global__ __launch_bounds__(512)
void gru_scan(const float* __restrict__ h0, const _Float16* __restrict__ Wh,
              const float* __restrict__ bh, const float* __restrict__ xproj,
              float* __restrict__ out) {
  __shared__ float hbuf[2][16][264];        // padded stride: fewer bank conflicts

  const int lane  = threadIdx.x & 31;
  const int wave  = threadIdx.x >> 5;       // 0..15
  const int lc    = lane & 15;
  const bool hi   = lane >= 16;
  const int mbase = blockIdx.x * 16;        // batch-row base
  const int c     = wave * 16 + lc;         // hidden column owned by this lane
  const int koff  = hi ? 16 : 0;

  for (int i = threadIdx.x; i < 16 * 256; i += 512) {
    const int mm = i >> 8, cc = i & 255;
    hbuf[0][mm][cc] = h0[(size_t)(mbase + mm) * H_ + cc];
  }

  // Register-resident recurrent weights: 3 gates x 8 k-blocks x v16h.
  v16h wR[8], wZ[8], wN[8];
#pragma unroll
  for (int kb = 0; kb < 8; ++kb) {
    wR[kb] = *(const v16h*)(Wh + (size_t)(c) * 256       + kb * 32 + koff);
    wZ[kb] = *(const v16h*)(Wh + (size_t)(256 + c) * 256 + kb * 32 + koff);
    wN[kb] = *(const v16h*)(Wh + (size_t)(512 + c) * 256 + kb * 32 + koff);
  }
  const float bhr = bh[c], bhz = bh[256 + c], bhn = bh[512 + c];
  __syncthreads();

  int cur = 0;
  for (int t = 0; t < T_; ++t) {
    // Prefetch next step's xproj lines (speculative; the one-past-the-end
    // prefetch at t==T_-1 is silently dropped per ISA).
#pragma unroll
    for (int vr = 0; vr < 8; ++vr) {
      const size_t xo1 = ((size_t)(mbase + vr + (hi ? 8 : 0)) * T_ + (t + 1)) * G_;
      __builtin_prefetch(&xproj[xo1 + c], 0, 3);
      __builtin_prefetch(&xproj[xo1 + 256 + c], 0, 3);
      __builtin_prefetch(&xproj[xo1 + 512 + c], 0, 3);
    }

    v8f accR = {}, accZ = {}, accN = {};
#pragma unroll
    for (int kb = 0; kb < 8; ++kb) {
      v16h a = make_a_frag(&hbuf[cur][lc][kb * 32 + (hi ? 16 : 0)]);
      accR = wmma_f16(a, wR[kb], accR);
      accZ = wmma_f16(a, wZ[kb], accZ);
      accN = wmma_f16(a, wN[kb], accN);
    }
#pragma unroll
    for (int vr = 0; vr < 8; ++vr) {
      const int mm = vr + (hi ? 8 : 0);
      const size_t xo = ((size_t)(mbase + mm) * T_ + t) * G_;
      const float gr = accR[vr] + xproj[xo + c]       + bhr;
      const float gz = accZ[vr] + xproj[xo + 256 + c] + bhz;
      const float an = accN[vr] + bhn;                         // h@W_hn^T + b_hn
      const float r  = fast_sigmoid(gr);
      const float z  = fast_sigmoid(gz);
      const float nt = fast_tanh(xproj[xo + 512 + c] + r * an);
      const float ho = hbuf[cur][mm][c];
      const float hn = (1.0f - z) * nt + z * ho;
      out[((size_t)(mbase + mm) * T_ + t) * H_ + c] = hn;
      hbuf[cur ^ 1][mm][c] = hn;              // write next-step buffer
    }
    __syncthreads();                          // single barrier per step
    cur ^= 1;
  }

  // final hidden state h_n, appended after (B,T,H) output
#pragma unroll
  for (int vr = 0; vr < 8; ++vr) {
    const int mm = vr + (hi ? 8 : 0);
    out[(size_t)B_ * T_ * H_ + (size_t)(mbase + mm) * H_ + c] = hbuf[cur][mm][c];
  }
}

extern "C" void kernel_launch(void* const* d_in, const int* in_sizes, int n_in,
                              void* d_out, int out_size, void* d_ws, size_t ws_size,
                              hipStream_t stream) {
  (void)in_sizes; (void)n_in; (void)out_size; (void)ws_size;
  const float* x    = (const float*)d_in[0];
  const float* h0   = (const float*)d_in[1];
  const float* Wirz = (const float*)d_in[2];
  const float* birz = (const float*)d_in[3];
  const float* Whrz = (const float*)d_in[4];
  const float* bhrz = (const float*)d_in[5];
  const float* Win  = (const float*)d_in[6];
  const float* bin  = (const float*)d_in[7];
  const float* Whn  = (const float*)d_in[8];
  const float* bhn  = (const float*)d_in[9];
  float* out = (float*)d_out;

  char* ws = (char*)d_ws;
  _Float16* Wx    = (_Float16*)(ws);
  _Float16* Wh    = (_Float16*)(ws + 393216);
  float*    bx    = (float*)(ws + 786432);
  float*    bhv   = (float*)(ws + 789504);
  float*    xproj = (float*)(ws + 792576);

  gru_pack<<<G_, 256, 0, stream>>>(Wirz, Win, Whrz, Whn, birz, bin, bhrz, bhn,
                                   Wx, Wh, bx, bhv);
  gru_xproj<<<(B_ * T_) / 16, 256, 0, stream>>>(x, Wx, bx, xproj);
  gru_scan<<<B_ / 16, 512, 0, stream>>>(h0, Wh, bhv, xproj, out);
}
